// RNN_77738908058070
// MI455X (gfx1250) — compile-verified, matrix operand order checked by reading, not run on hardware
//
#include <hip/hip_runtime.h>

typedef __attribute__((ext_vector_type(2)))  float    v2f;
typedef __attribute__((ext_vector_type(8)))  float    v8f;
typedef __attribute__((ext_vector_type(16))) _Float16 v16h;
typedef __attribute__((ext_vector_type(4)))  int      v4i;

static constexpr int HDIM = 128;   // input dim == hidden dim == 128
static constexpr int TT   = 2048;  // timesteps
static constexpr int BB   = 64;    // batch

__device__ __forceinline__ float fast_tanh(float x) {
  // tanh(x) = sign(x) * (1 - e) / (1 + e), e = exp(-2|x|)
  float ax = __builtin_fabsf(x);
  float e  = __expf(-2.0f * ax);
  float r  = (1.0f - e) * __builtin_amdgcn_rcpf(1.0f + e);
  return __builtin_copysignf(r, x);
}

// ---------------------------------------------------------------------------
// C[M x 128] = A[M x 128] @ W^T + bias1 (+ bias2)    (W row-major [128 x 128])
// fp32 WMMA 16x16x4. One workgroup = 128 rows, one wave = one 16-row strip.
// W staged to LDS via CDNA5 async global->LDS copies (ASYNCcnt tracked).
// ---------------------------------------------------------------------------
__global__ __launch_bounds__(256) void gemm128_kernel(
    const float* __restrict__ A, const float* __restrict__ W,
    const float* __restrict__ bias1, const float* __restrict__ bias2,
    float* __restrict__ C)
{
  __shared__ __align__(16) float ldsW[HDIM * HDIM];   // 64 KB of 320 KB WGP LDS
  const int tid = threadIdx.x;

#if __has_builtin(__builtin_amdgcn_global_load_async_to_lds_b128)
  {
    typedef __attribute__((address_space(1))) v4i glb_v4i;  // global src
    typedef __attribute__((address_space(3))) v4i lds_v4i;  // LDS dst
    glb_v4i* wg = (glb_v4i*)W;
    lds_v4i* wl = (lds_v4i*)ldsW;
    #pragma unroll
    for (int i = 0; i < 16; ++i)
      __builtin_amdgcn_global_load_async_to_lds_b128(wg + tid + i * 256,
                                                     wl + tid + i * 256, 0, 0);
  #if __has_builtin(__builtin_amdgcn_s_wait_asynccnt)
    __builtin_amdgcn_s_wait_asynccnt(0);
  #else
    asm volatile("s_wait_asynccnt 0" ::: "memory");
  #endif
  }
#else
  #pragma unroll
  for (int i = 0; i < 16; ++i)
    ((float4*)ldsW)[tid + i * 256] = ((const float4*)W)[tid + i * 256];
#endif
  __syncthreads();

  const int wave  = tid >> 5;
  const int lane  = tid & 31;
  const int n     = lane & 15;     // tile row/col within 16
  const int halfk = lane >> 4;     // K sub-pair select (16x16x4 f32 layout)
  const int row0  = blockIdx.x * 128 + wave * 16;

  // Preload this strip's A fragments: 32 K-chunks of 4, float2 per lane.
  // Layout (ISA 7.12.2, 32-bit A 16x4): lane m=lane&15, k0 = 4*kk + 2*halfk.
  v2f areg[32];
  const float* arow = A + (size_t)(row0 + n) * HDIM;
  #pragma unroll
  for (int kk = 0; kk < 32; ++kk)
    areg[kk] = *(const v2f*)(arow + kk * 4 + 2 * halfk);

  #pragma unroll 1
  for (int j = 0; j < 8; ++j) {           // 8 column tiles of 16
    const int ncol = j * 16 + n;
    // two independent accumulator chains (even/odd K-chunks) -> 2x WMMA ILP
    v8f acc0 = {0.f, 0.f, 0.f, 0.f, 0.f, 0.f, 0.f, 0.f};
    v8f acc1 = {0.f, 0.f, 0.f, 0.f, 0.f, 0.f, 0.f, 0.f};
    // B[k][ncol] = W[ncol][k]  -> consecutive-k float2 from LDS row `ncol`
    const float* wrow = ldsW + ncol * HDIM + 2 * halfk;
    #pragma unroll
    for (int kk = 0; kk < 32; kk += 2) {
      v2f b0 = *(const v2f*)(wrow + kk * 4);
      v2f b1 = *(const v2f*)(wrow + kk * 4 + 4);
      acc0 = __builtin_amdgcn_wmma_f32_16x16x4_f32(
          false, areg[kk], false, b0, (short)0, acc0, false, false);
      acc1 = __builtin_amdgcn_wmma_f32_16x16x4_f32(
          false, areg[kk + 1], false, b1, (short)0, acc1, false, false);
    }
    v8f acc = acc0 + acc1;
    float bv = bias1[ncol];
    if (bias2) bv += bias2[ncol];
    // C/D layout: lane holds col n, rows v + 8*halfk
    float* crow = C + (size_t)row0 * HDIM + ncol;
    #pragma unroll
    for (int v = 0; v < 8; ++v)
      crow[(size_t)(v + 8 * halfk) * HDIM] = acc[v] + bv;
  }
}

// ---------------------------------------------------------------------------
// Sequential scan: h_t = tanh(xp_t + h_{t-1} @ W_hh^T), per-WG 16 batch rows.
// f16 WMMA 16x16x32, W_hh^T B-fragments register-resident for all 2048 steps.
// h exchanged between waves through a double-buffered 2x4KB LDS buffer kept
// pre-swizzled in the WMMA A-fragment layout -> ONE barrier per step.
// ---------------------------------------------------------------------------
__global__ __launch_bounds__(256) void rnn_scan_kernel(
    const float* __restrict__ XP,    // [B][T][H] fp32 (x W_ih^T + b_ih + b_hh)
    const float* __restrict__ W_hh,  // [H][H] fp32
    float* __restrict__ HID)         // [B][T][H] fp32
{
  __shared__ __align__(32) _Float16 hbuf[2][4 * 32 * 16]; // [buf][kchunk][lane][16]

  const int tid   = threadIdx.x;
  const int wave  = tid >> 5;       // N-tile 0..7 (16 cols each)
  const int lane  = tid & 31;
  const int n     = lane & 15;
  const int halfk = lane >> 4;
  const int ncol  = wave * 16 + n;
  const int b0    = blockIdx.x * 16;

  // ---- B fragments: B[k][ncol] = W_hh[ncol][k], f16 16x16x32 layout ----
  // k' -> vgpr v: k'<8:v=k'/2 h=0; 8..15:v=(k'-8)/2 h=1; 16..23:v=4+.. h=0; 24..31 h=1
  v16h bf[4];
  #pragma unroll
  for (int c = 0; c < 4; ++c) {
    #pragma unroll
    for (int v = 0; v < 8; ++v) {
      int kbase = ((v < 4) ? 2 * v : 16 + 2 * (v - 4)) + 8 * halfk;
      v2f w = *(const v2f*)(W_hh + (size_t)ncol * HDIM + 32 * c + kbase);
      bf[c][2 * v]     = (_Float16)w.x;
      bf[c][2 * v + 1] = (_Float16)w.y;
    }
  }

  // ---- scatter offsets: element (m, k=ncol) into A-frag-layout hbuf ----
  const int kp    = ncol & 31;
  const int cW    = ncol >> 5;
  const int halfA = (kp >> 3) & 1;
  const int vA    = ((kp & 7) >> 1) + 4 * (kp >> 4);
  const int off0  = cW * 512 + 256 * halfA + 2 * vA + (kp & 1);

  // h0 = 0 (buffer 0 is read at t=0)
  #pragma unroll
  for (int i = 0; i < 8; ++i) hbuf[0][tid + i * 256] = (_Float16)0.f;

  // per-row global pointers (C-tile layout: lane owns col ncol, rows v+8*halfk)
  const float* xpp[8];
  float* hop[8];
  #pragma unroll
  for (int v = 0; v < 8; ++v) {
    int m = v + 8 * halfk;
    xpp[v] = XP  + ((size_t)(b0 + m) * TT) * HDIM + ncol;
    hop[v] = HID + ((size_t)(b0 + m) * TT) * HDIM + ncol;
  }
  __syncthreads();

  for (int t = 0; t < TT; ++t) {
    const _Float16* rb = hbuf[t & 1];         // h_{t-1}
    _Float16*       wb = hbuf[(t + 1) & 1];   // h_t (disjoint -> no mid barrier)

    // A fragments of h_{t-1}: contiguous 32B per lane per K-chunk
    v16h a0 = *(const v16h*)(rb + 0 * 512 + lane * 16);
    v16h a1 = *(const v16h*)(rb + 1 * 512 + lane * 16);
    v16h a2 = *(const v16h*)(rb + 2 * 512 + lane * 16);
    v16h a3 = *(const v16h*)(rb + 3 * 512 + lane * 16);

    float xpv[8];
    #pragma unroll
    for (int v = 0; v < 8; ++v) xpv[v] = xpp[v][t * HDIM];
    // hide next step's xp HBM latency behind this step's compute
    #pragma unroll
    for (int v = 0; v < 8; ++v) __builtin_prefetch(xpp[v] + (t + 1) * HDIM, 0, 0);

    // two independent accumulator chains -> WMMA chain depth 2
    v8f acc0 = {0.f, 0.f, 0.f, 0.f, 0.f, 0.f, 0.f, 0.f};
    v8f acc1 = {0.f, 0.f, 0.f, 0.f, 0.f, 0.f, 0.f, 0.f};
    acc0 = __builtin_amdgcn_wmma_f32_16x16x32_f16(false, a0, false, bf[0], (short)0, acc0, false, false);
    acc1 = __builtin_amdgcn_wmma_f32_16x16x32_f16(false, a1, false, bf[1], (short)0, acc1, false, false);
    acc0 = __builtin_amdgcn_wmma_f32_16x16x32_f16(false, a2, false, bf[2], (short)0, acc0, false, false);
    acc1 = __builtin_amdgcn_wmma_f32_16x16x32_f16(false, a3, false, bf[3], (short)0, acc1, false, false);

    #pragma unroll
    for (int v = 0; v < 8; ++v) {
      float hval = fast_tanh(acc0[v] + acc1[v] + xpv[v]);
      hop[v][t * HDIM] = hval;                       // fire-and-forget store
      int m = v + 8 * halfk;
      wb[off0 + 16 * m] = (_Float16)hval;            // pre-swizzled scatter
    }
    __syncthreads();  // h_t visible to all waves before step t+1 reads it
  }
}

// ---------------------------------------------------------------------------
extern "C" void kernel_launch(void* const* d_in, const int* in_sizes, int n_in,
                              void* d_out, int out_size, void* d_ws, size_t ws_size,
                              hipStream_t stream) {
  (void)in_sizes; (void)n_in; (void)out_size; (void)d_ws; (void)ws_size;
  const float* seq   = (const float*)d_in[0];
  const float* W_ih  = (const float*)d_in[1];
  const float* b_ih  = (const float*)d_in[2];
  const float* W_hh  = (const float*)d_in[3];
  const float* b_hh  = (const float*)d_in[4];
  const float* W_out = (const float*)d_in[5];
  const float* b_out = (const float*)d_in[6];

  float* out = (float*)d_out;
  const size_t NROW = (size_t)BB * TT;       // 131072 rows
  float* hidden = out;                       // outputs[0]: hidden_states
  float* outst  = out + NROW * HDIM;         // outputs[1]: output_states
  float* xproj  = outst;                     // stage x-proj in outputs[1] (overwritten later)

  dim3 blk(256);
  // 1) xp = seq @ W_ih^T + b_ih + b_hh       (memory-bound, fp32 WMMA)
  gemm128_kernel<<<dim3((unsigned)(NROW / 128)), blk, 0, stream>>>(
      seq, W_ih, b_ih, b_hh, xproj);
  // 2) recurrence: h_t = tanh(xp_t + h_{t-1} W_hh^T)   (latency-bound, f16 WMMA)
  rnn_scan_kernel<<<dim3(BB / 16), blk, 0, stream>>>(xproj, W_hh, hidden);
  // 3) out = hidden @ W_out^T + b_out        (memory-bound, fp32 WMMA)
  gemm128_kernel<<<dim3((unsigned)(NROW / 128)), blk, 0, stream>>>(
      hidden, W_out, b_out, nullptr, outst);
}
